// TSegNet_39444979647059
// MI455X (gfx1250) — matrix operand order repaired and checked by reading, not compiled, not run on hardware
//
#include <hip/hip_runtime.h>

// ---------------------------------------------------------------------------
// PointNet++ MSG segmentation (TSegNet) for gfx1250 (MI455X, wave32, WMMA).
// All MLP layers run through one fused WMMA GEMM kernel operating on
// zero-padded f16 operands (Kp = roundup32(K)), so the inner loop is
// 4x global_load_b128 + v_wmma_f32_16x16x32_f16 with no per-element guards.
// ---------------------------------------------------------------------------

typedef __attribute__((ext_vector_type(16))) _Float16 v16h;
typedef __attribute__((ext_vector_type(8)))  _Float16 v8h;
typedef __attribute__((ext_vector_type(8)))  float    v8f;

#define BN_INV_F 0.9999950000374997f

static __host__ __device__ inline int cdiv(int a, int b) { return (a + b - 1) / b; }
static inline int r16(int x) { return (x + 15) & ~15; }
static inline int r32(int x) { return (x + 31) & ~31; }

// ---------------------------------------------------------------------------
// (B,6,N) -> points (B,N,6) and xyz (B,N,3)
// ---------------------------------------------------------------------------
__global__ void transpose_in_kernel(const float* __restrict__ in,
                                    float* __restrict__ pts, float* __restrict__ xyz,
                                    int Bn, int Cn, int N) {
    long tot = (long)Bn * Cn * N;
    long gid = (long)blockIdx.x * blockDim.x + threadIdx.x;
    if (gid >= tot) return;
    int c = gid % Cn;
    int n = (gid / Cn) % N;
    int b = gid / ((long)Cn * N);
    float v = in[((size_t)b * Cn + c) * N + n];
    pts[((size_t)b * N + n) * Cn + c] = v;
    if (c < 3) xyz[((size_t)b * N + n) * 3 + c] = v;
}

// ---------------------------------------------------------------------------
// Farthest point sampling: one block per batch; per-thread distance chunks in
// registers, shared-mem argmax reduction. Emits pre-update far index.
// ---------------------------------------------------------------------------
template <int CH>
__global__ __launch_bounds__(256) void fps_kernel(const float* __restrict__ xyz,
                                                  int N, int S, int* __restrict__ out) {
    __shared__ float rv[256];
    __shared__ int   ri[256];
    __shared__ int   s_far;
    const int b = blockIdx.x;
    const int t = threadIdx.x;
    const float* xb = xyz + (size_t)b * N * 3;
    float dloc[CH];
#pragma unroll
    for (int j = 0; j < CH; ++j) dloc[j] = 1e10f;
    if (t == 0) s_far = 0;
    __syncthreads();
    for (int it = 0; it < S; ++it) {
        int far = s_far;
        if (t == 0) out[(size_t)b * S + it] = far;
        float cx = xb[(size_t)far * 3 + 0];
        float cy = xb[(size_t)far * 3 + 1];
        float cz = xb[(size_t)far * 3 + 2];
        float bv = -1.0f; int bi = 0;
#pragma unroll
        for (int j = 0; j < CH; ++j) {
            int i = j * 256 + t;
            if (i < N) {
                float dx = xb[(size_t)i * 3 + 0] - cx;
                float dy = xb[(size_t)i * 3 + 1] - cy;
                float dz = xb[(size_t)i * 3 + 2] - cz;
                float d  = dx * dx + dy * dy + dz * dz;
                float nd = fminf(dloc[j], d);
                dloc[j]  = nd;
                if (nd > bv) { bv = nd; bi = i; }
            }
        }
        rv[t] = bv; ri[t] = bi;
        __syncthreads();
        for (int s2 = 128; s2 > 0; s2 >>= 1) {
            if (t < s2) {
                float ov = rv[t + s2]; int oi = ri[t + s2];
                if (ov > rv[t] || (ov == rv[t] && oi < ri[t])) { rv[t] = ov; ri[t] = oi; }
            }
            __syncthreads();
        }
        if (t == 0) s_far = ri[0];
        __syncthreads();
    }
}

__global__ void gather_xyz_kernel(const float* __restrict__ xyz, const int* __restrict__ idx,
                                  int N, int S, int BS, float* __restrict__ out) {
    int gid = blockIdx.x * blockDim.x + threadIdx.x;
    if (gid >= BS) return;
    int b = gid / S;
    int j = idx[gid];
    const float* p = xyz + ((size_t)b * N + j) * 3;
    out[(size_t)gid * 3 + 0] = p[0];
    out[(size_t)gid * 3 + 1] = p[1];
    out[(size_t)gid * 3 + 2] = p[2];
}

// ---------------------------------------------------------------------------
// Ball query: first Ksamp points (ascending index) with d<=r^2; pad with
// first hit; if none, all = N-1 (matches reference clip semantics).
// ---------------------------------------------------------------------------
__global__ void ball_query_kernel(const float* __restrict__ xyz, const float* __restrict__ nxyz,
                                  int N, int S, int Ksamp, float r2, int BS,
                                  int* __restrict__ idx) {
    int gid = blockIdx.x * blockDim.x + threadIdx.x;
    if (gid >= BS) return;
    int b = gid / S;
    const float* xb = xyz + (size_t)b * N * 3;
    float cx = nxyz[(size_t)gid * 3 + 0];
    float cy = nxyz[(size_t)gid * 3 + 1];
    float cz = nxyz[(size_t)gid * 3 + 2];
    int* o = idx + (size_t)gid * Ksamp;
    int cnt = 0;
    for (int j = 0; j < N; ++j) {
        float dx = xb[(size_t)j * 3 + 0] - cx;
        float dy = xb[(size_t)j * 3 + 1] - cy;
        float dz = xb[(size_t)j * 3 + 2] - cz;
        float d = dx * dx + dy * dy + dz * dz;
        if (d <= r2) { o[cnt++] = j; if (cnt == Ksamp) break; }
    }
    if (cnt == 0) {
        for (int k = 0; k < Ksamp; ++k) o[k] = N - 1;
    } else {
        for (int k = cnt; k < Ksamp; ++k) o[k] = o[0];
    }
}

// g[(bs*K+k), 0..Kp) = [ points[b,idx,0..Cpts) , xyz[b,idx]-new_xyz[bs] , 0... ]  (f16)
__global__ void group_kernel(const float* __restrict__ xyz, const float* __restrict__ pts,
                             const float* __restrict__ nxyz, const int* __restrict__ idx,
                             int S, int Ksamp, int N, int Cpts, int BS, int Kp,
                             _Float16* __restrict__ g) {
    long tot = (long)BS * Ksamp * Kp;
    long gid = (long)blockIdx.x * blockDim.x + threadIdx.x;
    if (gid >= tot) return;
    int  c  = gid % Kp;
    long r  = gid / Kp;
    int  k  = r % Ksamp;
    int  bs = r / Ksamp;
    int  b  = bs / S;
    float v = 0.0f;
    if (c < Cpts + 3) {
        int j = idx[(size_t)bs * Ksamp + k];
        if (c < Cpts) v = pts[((size_t)b * N + j) * Cpts + c];
        else {
            int d = c - Cpts;
            v = xyz[((size_t)b * N + j) * 3 + d] - nxyz[(size_t)bs * 3 + d];
        }
    }
    g[gid] = (_Float16)v;
}

// f32 [N,K] weights -> f16 [Np,Kp], zero padded rows & cols
__global__ void pack_w_kernel(const float* __restrict__ w, _Float16* __restrict__ out,
                              int N, int K, int Np, int Kp) {
    int tot = Np * Kp;
    int gid = blockIdx.x * blockDim.x + threadIdx.x;
    if (gid >= tot) return;
    int c = gid % Kp, r = gid / Kp;
    out[gid] = (r < N && c < K) ? (_Float16)w[(size_t)r * K + c] : (_Float16)0.0f;
}

// ---------------------------------------------------------------------------
// WMMA GEMM: Y = act(Xp[M,Kp] @ Wp[Np,Kp]^T + b).
// Xp/Wp are zero-padded f16; Kp % 32 == 0; M % 16 == 0; Wp rows >= Npad.
// mode 0: f32 out, y = acc + b            (write cols < N, row stride N)
// mode 1: f32 out, y = relu((acc+b)*gs+bt)(write cols < N, row stride N)
// mode 2: f16 out, bn+relu, write ALL Npad cols (zeros beyond N -> next Kp pad)
// ---------------------------------------------------------------------------
__global__ __launch_bounds__(256) void gemm_wmma_kernel(
    const _Float16* __restrict__ Xp, const _Float16* __restrict__ Wp,
    const float* __restrict__ bias, const float* __restrict__ gamma,
    const float* __restrict__ beta, void* __restrict__ Yv,
    int M, int Kp, int N, int Npad, int mode) {
    const int Mt  = M >> 4;
    const int Ntc = Npad >> 4;
    const int gw = blockIdx.x * 8 + (threadIdx.x >> 5);   // uniform per wave
    if (gw >= Mt * Ntc) return;
    const int mt   = gw / Ntc;
    const int nt   = gw % Ntc;
    const int lane = threadIdx.x & 31;
    const int half = lane >> 4;
    const int l15  = lane & 15;

    const int row = mt * 16 + l15;
    const int col = nt * 16 + l15;
    const _Float16* xr = Xp + (size_t)row * Kp;
    const _Float16* wr = Wp + (size_t)col * Kp;

    v8f acc = {};
    for (int k0 = 0; k0 < Kp; k0 += 32) {
        // A 16x32 f16: lanes 0-15 row=l15, K = k0+{0..7,16..23}; lanes 16-31 K = k0+{8..15,24..31}
        const int ka = k0 + (half ? 8 : 0);
        v8h alo = *(const v8h*)(xr + ka);
        v8h ahi = *(const v8h*)(xr + ka + 16);
        // B 32x16 f16: lanes 0-15 col=l15, K = k0+{0..7,16..23}; lanes 16-31 K = k0+{8..15,24..31}
        const int kb = k0 + (half ? 8 : 0);
        v8h blo = *(const v8h*)(wr + kb);
        v8h bhi = *(const v8h*)(wr + kb + 16);
        __builtin_prefetch(xr + ka + 32, 0, 1);
        __builtin_prefetch(wr + kb + 32, 0, 1);
        v16h a  = __builtin_shufflevector(alo, ahi, 0, 1, 2, 3, 4, 5, 6, 7,
                                          8, 9, 10, 11, 12, 13, 14, 15);
        v16h bf = __builtin_shufflevector(blo, bhi, 0, 1, 2, 3, 4, 5, 6, 7,
                                          8, 9, 10, 11, 12, 13, 14, 15);
        acc = __builtin_amdgcn_wmma_f32_16x16x32_f16(
            false, a, false, bf, (short)0, acc, false, false);
    }

    const bool cok = col < N;
    float bb = cok ? bias[col] : 0.0f;
    float gs = 1.0f, bt = 0.0f;
    if (mode >= 1) {
        gs = cok ? gamma[col] * BN_INV_F : 0.0f;
        bt = cok ? beta[col] : 0.0f;
    }
    const int rbase = mt * 16 + (half ? 8 : 0);
    if (mode == 2) {
        _Float16* Yh = (_Float16*)Yv;
#pragma unroll
        for (int r = 0; r < 8; ++r) {
            float v = acc[r] + bb;
            v = v * gs + bt;
            v = v > 0.0f ? v : 0.0f;
            Yh[(size_t)(rbase + r) * Npad + col] = (_Float16)v;
        }
    } else {
        if (!cok) return;
        float* Yf = (float*)Yv;
#pragma unroll
        for (int r = 0; r < 8; ++r) {
            float v = acc[r] + bb;
            if (mode == 1) { v = v * gs + bt; v = v > 0.0f ? v : 0.0f; }
            Yf[(size_t)(rbase + r) * N + col] = v;
        }
    }
}

// out[bs, coff+c] = max_k in[(bs*K+k), c]
__global__ void maxpool_kernel(const float* __restrict__ in, float* __restrict__ out,
                               int BS, int Ksamp, int C, int Ctot, int coff) {
    int gid = blockIdx.x * blockDim.x + threadIdx.x;
    if (gid >= BS * C) return;
    int bs = gid / C, c = gid % C;
    const float* p = in + (size_t)bs * Ksamp * C + c;
    float m = p[0];
    for (int k = 1; k < Ksamp; ++k) m = fmaxf(m, p[(size_t)k * C]);
    out[(size_t)bs * Ctot + coff + c] = m;
}

// ---------------------------------------------------------------------------
// FP: 3-NN inverse-distance interp + concat [points1 | interp(points2)] -> f16 padded
// ---------------------------------------------------------------------------
__global__ void interp_concat_kernel(const float* __restrict__ xyz1, const float* __restrict__ xyz2,
                                     const float* __restrict__ p1, const float* __restrict__ p2,
                                     int Bn, int N1, int N2, int C1, int C2, int Kp,
                                     _Float16* __restrict__ out) {
    int gid = blockIdx.x * blockDim.x + threadIdx.x;
    if (gid >= Bn * N1) return;
    int b = gid / N1;
    const float* q = xyz1 + (size_t)gid * 3;
    float px = q[0], py = q[1], pz = q[2];
    const float* x2 = xyz2 + (size_t)b * N2 * 3;
    float d0 = 1e30f, d1 = 1e30f, d2 = 1e30f;
    int   i0 = 0,     i1 = 0,     i2 = 0;
    for (int j = 0; j < N2; ++j) {
        float dx = x2[(size_t)j * 3 + 0] - px;
        float dy = x2[(size_t)j * 3 + 1] - py;
        float dz = x2[(size_t)j * 3 + 2] - pz;
        float d = dx * dx + dy * dy + dz * dz;
        if (d < d0)      { d2 = d1; i2 = i1; d1 = d0; i1 = i0; d0 = d; i0 = j; }
        else if (d < d1) { d2 = d1; i2 = i1; d1 = d;  i1 = j; }
        else if (d < d2) { d2 = d;  i2 = j; }
    }
    float w0 = 1.0f / (d0 + 1e-8f);
    float w1 = 1.0f / (d1 + 1e-8f);
    float w2 = 1.0f / (d2 + 1e-8f);
    float ws = w0 + w1 + w2;
    w0 /= ws; w1 /= ws; w2 /= ws;
    _Float16* o = out + (size_t)gid * Kp;
    const float* r1 = p1 + (size_t)gid * C1;
    for (int c = 0; c < C1; ++c) o[c] = (_Float16)r1[c];
    const float* base = p2 + (size_t)b * N2 * C2;
    const float* a0 = base + (size_t)i0 * C2;
    const float* a1 = base + (size_t)i1 * C2;
    const float* a2 = base + (size_t)i2 * C2;
    for (int c = 0; c < C2; ++c)
        o[C1 + c] = (_Float16)(w0 * a0[c] + w1 * a1[c] + w2 * a2[c]);
    for (int c = C1 + C2; c < Kp; ++c) o[c] = (_Float16)0.0f;
}

// (B,N,C) -> (B,C,N)
__global__ void transpose_out_kernel(const float* __restrict__ in, float* __restrict__ out,
                                     int Bn, int N, int C) {
    long tot = (long)Bn * N * C;
    long gid = (long)blockIdx.x * blockDim.x + threadIdx.x;
    if (gid >= tot) return;
    int c = gid % C;
    int n = (gid / C) % N;
    int b = gid / ((long)C * N);
    out[((size_t)b * C + c) * N + n] = in[gid];
}

// ---------------------------------------------------------------------------
// Host side
// ---------------------------------------------------------------------------
struct BnP { const float *w, *b, *g, *beta; };

static inline BnP take_bn(void* const* d_in, int& p) {
    BnP r{ (const float*)d_in[p], (const float*)d_in[p + 1],
           (const float*)d_in[p + 2], (const float*)d_in[p + 3] };
    p += 4;
    return r;
}

// pack weights + run one fused GEMM layer
static void launch_gemm(const _Float16* Xp, const float* W, const float* b,
                        const float* g, const float* beta, void* Y,
                        int M, int K, int N, int mode, _Float16* Wbuf,
                        hipStream_t stream) {
    int Kp = r32(K);
    int Np = r32(N);
    pack_w_kernel<<<cdiv(Np * Kp, 256), 256, 0, stream>>>(W, Wbuf, N, K, Np, Kp);
    int Npad = (mode == 2) ? Np : r16(N);
    int T = (M >> 4) * (Npad >> 4);
    gemm_wmma_kernel<<<cdiv(T, 8), 256, 0, stream>>>(Xp, Wbuf, b, g, beta, Y,
                                                     M, Kp, N, Npad, mode);
}

extern "C" void kernel_launch(void* const* d_in, const int* in_sizes, int n_in,
                              void* d_out, int out_size, void* d_ws, size_t ws_size,
                              hipStream_t stream) {
    (void)in_sizes; (void)n_in; (void)out_size; (void)ws_size;
    const int B = 4, N0 = 16384;
    const int S1 = 1024, S2 = 512, S3 = 256;

    // ---- workspace layout ----
    float* ws = (float*)d_ws;
    size_t off = 0;
    auto alloc = [&](size_t n) { float* p = ws + off; off += (n + 7) & ~(size_t)7; return p; };
    auto alloch = [&](size_t nh) { return (_Float16*)alloc((nh + 1) / 2); };

    float* P0     = alloc((size_t)B * N0 * 6);     // l0_points (B,N,6) f32
    float* X0     = alloc((size_t)B * N0 * 3);     // l0_xyz
    float* L1X    = alloc((size_t)B * S1 * 3);
    float* L2X    = alloc((size_t)B * S2 * 3);
    float* L3X    = alloc((size_t)B * S3 * 3);
    float* L1P    = alloc((size_t)B * S1 * 64);
    float* L2P    = alloc((size_t)B * S2 * 256);
    float* L3P    = alloc((size_t)B * S3 * 512);
    float* L2PN   = alloc((size_t)B * S2 * 256);   // fp3 out (f32: interp consumer)
    float* L1PN   = alloc((size_t)B * S1 * 128);   // fp2 out
    float* F32S   = alloc((size_t)8400000);        // f32 GEMM outputs (maxpool/seg/edge)
    _Float16* GA16   = alloch((size_t)10500000);   // f16 ping
    _Float16* GB16   = alloch((size_t)8400000);    // f16 pong
    _Float16* L0PN16 = alloch((size_t)B * N0 * 64);// fp1 out, f16 padded (head input)
    _Float16* W16    = alloch((size_t)262144);     // packed weights
    int* FPSI = (int*)alloc((size_t)B * 1024);
    int* BIDX = (int*)alloc((size_t)B * 1024 * 32);

    // ---- params (setup_inputs insertion order; conv_bn = w,b,g,beta; lin = w,b) ----
    int p = 1;  // d_in[0] = xyz
    BnP sa[3][2][3];
    for (int i = 0; i < 3; ++i)
        for (int br = 0; br < 2; ++br)
            for (int l = 0; l < 3; ++l) sa[i][br][l] = take_bn(d_in, p);
    BnP fp3l[2], fp2l[2], fp1l[2];
    for (int l = 0; l < 2; ++l) fp3l[l] = take_bn(d_in, p);
    for (int l = 0; l < 2; ++l) fp2l[l] = take_bn(d_in, p);
    for (int l = 0; l < 2; ++l) fp1l[l] = take_bn(d_in, p);
    BnP head = take_bn(d_in, p);
    const float* segw  = (const float*)d_in[p++];
    const float* segb  = (const float*)d_in[p++];
    const float* edgew = (const float*)d_in[p++];
    const float* edgeb = (const float*)d_in[p++];

    // ---- l0 ----
    {
        long tot = (long)B * 6 * N0;
        transpose_in_kernel<<<cdiv((int)tot, 256), 256, 0, stream>>>(
            (const float*)d_in[0], P0, X0, B, 6, N0);
    }

    // ---- SA stages ----
    const float radii[3][2]  = {{2.5f, 5.0f}, {5.0f, 10.0f}, {10.0f, 20.0f}};
    const int   ksamp[2]     = {16, 32};
    const int   dims[3][4]   = {{9, 9, 32, 32}, {67, 64, 64, 128}, {259, 256, 196, 256}};
    const int   ctot[3]      = {64, 256, 512};
    const int   sin_arr[3]   = {N0, S1, S2};
    const int   sout_arr[3]  = {S1, S2, S3};
    const float* xin_arr[3]  = {X0, L1X, L2X};
    const float* pin_arr[3]  = {P0, L1P, L2P};
    const int    cpts_arr[3] = {6, 64, 256};
    float* xout_arr[3] = {L1X, L2X, L3X};
    float* pout_arr[3] = {L1P, L2P, L3P};

    for (int st = 0; st < 3; ++st) {
        const int Sin = sin_arr[st], Sout = sout_arr[st], Cpts = cpts_arr[st];
        const float* xin = xin_arr[st];
        const float* pin = pin_arr[st];
        float* xout = xout_arr[st];
        float* pout = pout_arr[st];

        if (Sin == 16384)     fps_kernel<64><<<B, 256, 0, stream>>>(xin, Sin, Sout, FPSI);
        else if (Sin == 1024) fps_kernel<4><<<B, 256, 0, stream>>>(xin, Sin, Sout, FPSI);
        else                  fps_kernel<2><<<B, 256, 0, stream>>>(xin, Sin, Sout, FPSI);

        int BS = B * Sout;
        gather_xyz_kernel<<<cdiv(BS, 256), 256, 0, stream>>>(xin, FPSI, Sin, Sout, BS, xout);

        int coff = 0;
        for (int br = 0; br < 2; ++br) {
            const int Kc = ksamp[br];
            const float r2 = radii[st][br] * radii[st][br];
            ball_query_kernel<<<cdiv(BS, 128), 128, 0, stream>>>(
                xin, xout, Sin, Sout, Kc, r2, BS, BIDX);
            const int rows = BS * Kc;
            const int* d = dims[st];
            const int Kp0 = r32(d[0]);
            long tot = (long)rows * Kp0;
            group_kernel<<<cdiv((int)tot, 256), 256, 0, stream>>>(
                xin, pin, xout, BIDX, Sout, Kc, Sin, Cpts, BS, Kp0, GA16);
            launch_gemm(GA16, sa[st][br][0].w, sa[st][br][0].b, sa[st][br][0].g, sa[st][br][0].beta,
                        GB16, rows, d[0], d[1], 2, W16, stream);
            launch_gemm(GB16, sa[st][br][1].w, sa[st][br][1].b, sa[st][br][1].g, sa[st][br][1].beta,
                        GA16, rows, r32(d[1]), d[2], 2, W16, stream);
            launch_gemm(GA16, sa[st][br][2].w, sa[st][br][2].b, sa[st][br][2].g, sa[st][br][2].beta,
                        F32S, rows, r32(d[2]), d[3], 1, W16, stream);
            maxpool_kernel<<<cdiv(BS * d[3], 256), 256, 0, stream>>>(
                F32S, pout, BS, Kc, d[3], ctot[st], coff);
            coff += d[3];
        }
    }

    // ---- FP stages ----
    struct FpDesc {
        const float *x1, *x2, *p1, *p2; int N1, N2, C1, C2, d1, d2; BnP* L;
        void* out; int outmode;
    } fpst[3] = {
        { L2X, L3X, L2P, L3P,  S2, S3, 256, 512, 256, 256, fp3l, (void*)L2PN,   1 },
        { L1X, L2X, L1P, L2PN, S1, S2,  64, 256, 128, 128, fp2l, (void*)L1PN,   1 },
        { X0,  L1X, P0,  L1PN, N0, S1,   6, 128,  64,  64, fp1l, (void*)L0PN16, 2 },
    };
    for (int st = 0; st < 3; ++st) {
        FpDesc& f = fpst[st];
        int BN1 = B * f.N1;
        int Kp = r32(f.C1 + f.C2);
        interp_concat_kernel<<<cdiv(BN1, 128), 128, 0, stream>>>(
            f.x1, f.x2, f.p1, f.p2, B, f.N1, f.N2, f.C1, f.C2, Kp, GA16);
        launch_gemm(GA16, f.L[0].w, f.L[0].b, f.L[0].g, f.L[0].beta, GB16,
                    BN1, f.C1 + f.C2, f.d1, 2, W16, stream);
        launch_gemm(GB16, f.L[1].w, f.L[1].b, f.L[1].g, f.L[1].beta, f.out,
                    BN1, f.d1, f.d2, f.outmode, W16, stream);
    }

    // ---- head ----
    const int MN = B * N0;
    // h (f16, Npad = r32(36) = 64, matches seg/edge Kp=r32(36)=64 with zero-padded weights)
    launch_gemm(L0PN16, head.w, head.b, head.g, head.beta, GA16, MN, 64, 36, 2, W16, stream);

    float* out_f = (float*)d_out;
    launch_gemm(GA16, segw, segb, nullptr, nullptr, F32S, MN, 36, 17, 0, W16, stream);
    {
        long tot = (long)MN * 17;
        transpose_out_kernel<<<cdiv((int)tot, 256), 256, 0, stream>>>(F32S, out_f, B, N0, 17);
    }
    launch_gemm(GA16, edgew, edgeb, nullptr, nullptr, F32S, MN, 36, 2, 0, W16, stream);
    {
        long tot = (long)MN * 2;
        transpose_out_kernel<<<cdiv((int)tot, 256), 256, 0, stream>>>(
            F32S, out_f + (size_t)B * 17 * N0, B, N0, 2);
    }
}
